// RGCN_37263136260546
// MI455X (gfx1250) — matrix operand order repaired and checked by reading, not compiled, not run on hardware
//
#include <hip/hip_runtime.h>
#include <math.h>

// ---------------------------------------------------------------------------
// RGCN fused kernels for gfx1250 (MI455X).
//  h = entity[node_ids]                               [N,128]
//  S[dst,rel] += h[src]  (atomic f32)                 [N,16,128]
//  out = sigmoid(relu(S/deg @ W + h @ Wself) @ fc_w + fc_b)
//  GEMMs: v_wmma_f32_16x16x32_bf16 (bf16 in, f32 accumulate).
//  B-tile staging: global_load_async_to_lds_b128 (ASYNCcnt).
// ---------------------------------------------------------------------------

typedef __attribute__((ext_vector_type(16))) __bf16 v16bf;
typedef __attribute__((ext_vector_type(8)))  __bf16 v8bf;
typedef __attribute__((ext_vector_type(2)))  __bf16 v2bf;
typedef __attribute__((ext_vector_type(8)))  float  v8f;
typedef __attribute__((ext_vector_type(4)))  int    v4i;

#define D_DIM 128
#define R_REL 16
#define K_S   2048   // R_REL * D_DIM
#define K_TOT 2176   // K_S + D_DIM (self-loop concatenated)
#define T_OUT 50
#define T_PAD 64
#define MT    32     // M tile per block

__device__ __forceinline__ unsigned short bf16_bits(float f) {
  unsigned u = __builtin_bit_cast(unsigned, f);
  return (unsigned short)((u + 0x7fffu + ((u >> 16) & 1u)) >> 16);  // RNE
}
__device__ __forceinline__ __bf16 to_bf16(float f) {
  unsigned short b = bf16_bits(f);
  return __builtin_bit_cast(__bf16, b);
}
// packed f32x2 -> bf16x2 (hardware v_cvt_pk_bf16_f32 when available)
__device__ __forceinline__ unsigned pack_bf16x2(float x, float y) {
#if defined(__gfx1250__) && __has_builtin(__builtin_amdgcn_cvt_pk_bf16_f32)
  v2bf p = __builtin_amdgcn_cvt_pk_bf16_f32(x, y);
  return __builtin_bit_cast(unsigned, p);
#else
  return (unsigned)bf16_bits(x) | ((unsigned)bf16_bits(y) << 16);
#endif
}

__device__ __forceinline__ v16bf make16(v8bf lo, v8bf hi) {
  return __builtin_shufflevector(lo, hi, 0, 1, 2, 3, 4, 5, 6, 7, 8, 9, 10, 11,
                                 12, 13, 14, 15);
}
// A fragment: elems 0-15 -> K akb..akb+15 (contiguous)
__device__ __forceinline__ v16bf fragA(const __bf16* rowbase, int akb) {
  v8bf lo = *(const v8bf*)(rowbase + akb);
  v8bf hi = *(const v8bf*)(rowbase + akb + 8);
  return make16(lo, hi);
}
// B fragment: elems 0-15 -> 16 contiguous K values
__device__ __forceinline__ v16bf fragB(const __bf16* p) {
  v8bf lo = *(const v8bf*)p;
  v8bf hi = *(const v8bf*)(p + 8);
  return make16(lo, hi);
}

// ---- async global->LDS copy (CDNA5 ASYNCcnt path), with safe fallback -----
__device__ __forceinline__ void async_copy_b128(const __bf16* gsrc,
                                                __bf16* lds_dst) {
#if defined(__gfx1250__) && \
    __has_builtin(__builtin_amdgcn_global_load_async_to_lds_b128)
  __builtin_amdgcn_global_load_async_to_lds_b128(
      (__attribute__((address_space(1))) v4i*)gsrc,
      (__attribute__((address_space(3))) v4i*)lds_dst, 0, 0);
#else
  *(uint4*)lds_dst = *(const uint4*)gsrc;
#endif
}
__device__ __forceinline__ void async_wait_all() {
#if defined(__gfx1250__) && __has_builtin(__builtin_amdgcn_s_wait_asynccnt)
  __builtin_amdgcn_s_wait_asynccnt(0);
#elif defined(__gfx1250__) && \
    __has_builtin(__builtin_amdgcn_global_load_async_to_lds_b128)
  asm volatile("s_wait_asynccnt 0x0" ::: "memory");
#endif
}

// ------------------------------ gather h -----------------------------------
__global__ void k_gather(const int* __restrict__ ids,
                         const float* __restrict__ ent, float* __restrict__ h,
                         int N) {
  long long tid = (long long)blockIdx.x * blockDim.x + threadIdx.x;  // N*32
  int n = (int)(tid >> 5), q = (int)(tid & 31);
  if (n >= N) return;
  float4 v = ((const float4*)(ent + (size_t)ids[n] * D_DIM))[q];
  ((float4*)(h + (size_t)n * D_DIM))[q] = v;
}

// --------------------- weight transpose + bf16 convert ---------------------
// WT[o][k] = (k < K_S) ? weight[k/128][k%128][o] : wself[k-K_S][o]
__global__ void k_cvt_w(const float* __restrict__ w,
                        const float* __restrict__ wself,
                        __bf16* __restrict__ WT) {
  int tid = blockIdx.x * blockDim.x + threadIdx.x;
  if (tid >= D_DIM * K_TOT) return;
  int o = tid / K_TOT, k = tid % K_TOT;
  float v = (k < K_S) ? w[(size_t)k * D_DIM + o]
                      : wself[(size_t)(k - K_S) * D_DIM + o];
  WT[(size_t)o * K_TOT + k] = to_bf16(v);
}

// FT[t][d] = fc_w[d][t], zero-padded to 64 rows
__global__ void k_cvt_fc(const float* __restrict__ fcw,
                         __bf16* __restrict__ FT) {
  int tid = blockIdx.x * blockDim.x + threadIdx.x;
  if (tid >= T_PAD * D_DIM) return;
  int t = tid >> 7, d = tid & 127;
  float v = (t < T_OUT) ? fcw[(size_t)d * T_OUT + t] : 0.0f;
  FT[(size_t)t * D_DIM + d] = to_bf16(v);
}

// ---------------------- edge scatter (atomic f32) --------------------------
__global__ void k_scatter(const int* __restrict__ src,
                          const int* __restrict__ dst,
                          const int* __restrict__ et,
                          const float* __restrict__ h, float* __restrict__ S,
                          float* __restrict__ deg, int E) {
  long long tid = (long long)blockIdx.x * blockDim.x + threadIdx.x;  // E*32
  int e = (int)(tid >> 5), q = (int)(tid & 31);
  if (e >= E) return;
  int s = src[e], d = dst[e], r = et[e];
  float4 v = ((const float4*)(h + (size_t)s * D_DIM))[q];
  float* p = S + ((size_t)d * R_REL + r) * D_DIM + q * 4;
  __hip_atomic_fetch_add(p + 0, v.x, __ATOMIC_RELAXED, __HIP_MEMORY_SCOPE_AGENT);
  __hip_atomic_fetch_add(p + 1, v.y, __ATOMIC_RELAXED, __HIP_MEMORY_SCOPE_AGENT);
  __hip_atomic_fetch_add(p + 2, v.z, __ATOMIC_RELAXED, __HIP_MEMORY_SCOPE_AGENT);
  __hip_atomic_fetch_add(p + 3, v.w, __ATOMIC_RELAXED, __HIP_MEMORY_SCOPE_AGENT);
  if (q == 0)
    __hip_atomic_fetch_add(deg + d, 1.0f, __ATOMIC_RELAXED,
                           __HIP_MEMORY_SCOPE_AGENT);
}

// ------------------- fused GEMM + self-loop + relu + FC --------------------
// Block = 256 threads (8 wave32). Computes rows [n0, n0+32).
// Wave w owns output cols [16w, 16w+16) with two M-tiles (2 accumulators).
__global__ __launch_bounds__(256, 2) void k_rgcn_gemm(
    const float* __restrict__ S, const float* __restrict__ h,
    const float* __restrict__ deg, const __bf16* __restrict__ WT,
    const __bf16* __restrict__ FT, const float* __restrict__ fc_b,
    float* __restrict__ out, int N) {
  __shared__ __bf16 As[MT * 32];        // A tile, bf16, deg-scaled
  __shared__ __bf16 Bs[D_DIM * 32];     // B tile (async-staged)
  __shared__ __bf16 Rs[MT * D_DIM];     // relu intermediate for FC
  __shared__ float invdeg[MT];

  const int tid = threadIdx.x;
  const int wave = tid >> 5;
  const int lane = tid & 31;
  const int n0 = blockIdx.x * MT;

  if (tid < MT) {
    float dg = (n0 + tid < N) ? deg[n0 + tid] : 1.0f;
    invdeg[tid] = 1.0f / fmaxf(dg, 1.0f);
  }

  // ---- staging maps (per-thread, loop-invariant)
  const int arow = tid >> 3;                 // 0..31
  const int acol = (tid & 7) * 4;            // 0..28 step 4
  const float* aS = S + (size_t)(n0 + arow) * K_S + acol;
  const float* aH = h + (size_t)(n0 + arow) * D_DIM + acol;
  __bf16* adst = As + arow * 32 + acol;

  const int brow = tid >> 1;                 // 0..127
  const int bcol = (tid & 1) * 16;           // 0 or 16
  const __bf16* bsrc = WT + (size_t)brow * K_TOT + bcol;
  __bf16* bdst = Bs + brow * 32 + bcol;

  // ---- WMMA fragment maps (contiguous k = 16*half + s, same for A and B)
  const int mrow = lane & 15;
  const int lohi = lane >> 4;
  const int akb = lohi * 16;                 // A: {0-15} / {16-31}
  const int bkb = lohi * 16;                 // B: {0-15} / {16-31}
  const __bf16* Bfrag = Bs + (wave * 16 + mrow) * 32 + bkb;
  const __bf16* Afrag0 = As + mrow * 32;
  const __bf16* Afrag1 = As + (16 + mrow) * 32;

  v8f acc0 = {}, acc1 = {};
  __syncthreads();
  const float sc = invdeg[arow];

  // ================= phase 1: relation block (deg-scaled S) ================
  for (int k0 = 0; k0 < K_S; k0 += 32) {
    async_copy_b128(bsrc + k0, bdst);        // B tile -> LDS (ASYNCcnt)
    async_copy_b128(bsrc + k0 + 8, bdst + 8);
    float4 v = *(const float4*)(aS + k0);    // A tile: f32 -> scale -> bf16
    if (k0 + 32 < K_S) __builtin_prefetch(aS + k0 + 32, 0, 1);
    uint2 pk;
    pk.x = pack_bf16x2(v.x * sc, v.y * sc);
    pk.y = pack_bf16x2(v.z * sc, v.w * sc);
    *(uint2*)adst = pk;
    async_wait_all();
    __syncthreads();

    v16bf b = fragB(Bfrag);
    v16bf a0 = fragA(Afrag0, akb);
    v16bf a1 = fragA(Afrag1, akb);
    acc0 = __builtin_amdgcn_wmma_f32_16x16x32_bf16(false, a0, false, b,
                                                   (short)0, acc0, false, false);
    acc1 = __builtin_amdgcn_wmma_f32_16x16x32_bf16(false, a1, false, b,
                                                   (short)0, acc1, false, false);
    __syncthreads();
  }

  // ================= phase 2: self-loop block (h, unscaled) ================
  for (int k0 = 0; k0 < D_DIM; k0 += 32) {
    async_copy_b128(bsrc + K_S + k0, bdst);
    async_copy_b128(bsrc + K_S + k0 + 8, bdst + 8);
    float4 v = *(const float4*)(aH + k0);
    uint2 pk;
    pk.x = pack_bf16x2(v.x, v.y);
    pk.y = pack_bf16x2(v.z, v.w);
    *(uint2*)adst = pk;
    async_wait_all();
    __syncthreads();

    v16bf b = fragB(Bfrag);
    v16bf a0 = fragA(Afrag0, akb);
    v16bf a1 = fragA(Afrag1, akb);
    acc0 = __builtin_amdgcn_wmma_f32_16x16x32_bf16(false, a0, false, b,
                                                   (short)0, acc0, false, false);
    acc1 = __builtin_amdgcn_wmma_f32_16x16x32_bf16(false, a1, false, b,
                                                   (short)0, acc1, false, false);
    __syncthreads();
  }

  // ================= relu -> LDS (bf16) ====================================
  {
    const int cc = wave * 16 + mrow;
    const int rbase = lohi * 8;
#pragma unroll
    for (int j = 0; j < 8; ++j)
      Rs[(rbase + j) * D_DIM + cc] = to_bf16(fmaxf(acc0[j], 0.0f));
#pragma unroll
    for (int j = 0; j < 8; ++j)
      Rs[(16 + rbase + j) * D_DIM + cc] = to_bf16(fmaxf(acc1[j], 0.0f));
  }
  __syncthreads();

  // ================= FC: [32x128] @ [128x50->64], all 8 waves ==============
  {
    const int mbase = (wave >> 2) * 16;      // waves 0-3: rows 0-15, 4-7: 16-31
    const int t0 = (wave & 3) * 16;          // output-class tile
    const __bf16* frow = FT + (size_t)(t0 + mrow) * D_DIM;
    const __bf16* rrow = Rs + (size_t)(mbase + mrow) * D_DIM;
    v8f acc2 = {};
#pragma unroll
    for (int k0 = 0; k0 < D_DIM; k0 += 32) {
      v16bf a = fragA(rrow + k0, akb);
      v16bf b = fragB(frow + k0 + bkb);
      acc2 = __builtin_amdgcn_wmma_f32_16x16x32_bf16(false, a, false, b,
                                                     (short)0, acc2, false,
                                                     false);
    }
    const int t = t0 + mrow;
    if (t < T_OUT) {
      const float bias = fc_b[t];
      const int rbase = lohi * 8;
#pragma unroll
      for (int j = 0; j < 8; ++j) {
        int row = n0 + mbase + rbase + j;
        if (row < N) {
          float x = acc2[j] + bias;
          out[(size_t)row * T_OUT + t] = 1.0f / (1.0f + expf(-x));
        }
      }
    }
  }
}

// ---------------------------------------------------------------------------
extern "C" void kernel_launch(void* const* d_in, const int* in_sizes, int n_in,
                              void* d_out, int out_size, void* d_ws,
                              size_t ws_size, hipStream_t stream) {
  const int* node_ids = (const int*)d_in[0];
  const int* src_idx = (const int*)d_in[1];
  const int* dst_idx = (const int*)d_in[2];
  const int* etype = (const int*)d_in[3];
  const float* entity = (const float*)d_in[4];
  const float* weight = (const float*)d_in[5];
  const float* wself = (const float*)d_in[6];
  const float* fc_w = (const float*)d_in[7];
  const float* fc_b = (const float*)d_in[8];
  const int N = in_sizes[0];
  const int E = in_sizes[1];

  // workspace layout (256B aligned)
  char* ws = (char*)d_ws;
  size_t offS = 0;
  size_t offH = offS + (size_t)N * K_S * sizeof(float);
  size_t offDg = offH + (size_t)N * D_DIM * sizeof(float);
  size_t offWT = offDg + (((size_t)N * sizeof(float) + 255) & ~(size_t)255);
  size_t offFT = offWT + (size_t)D_DIM * K_TOT * sizeof(unsigned short);
  float* Sbuf = (float*)(ws + offS);
  float* hbuf = (float*)(ws + offH);
  float* degb = (float*)(ws + offDg);
  __bf16* WT = (__bf16*)(ws + offWT);
  __bf16* FT = (__bf16*)(ws + offFT);
  float* outp = (float*)d_out;

  (void)hipMemsetAsync(Sbuf, 0, (size_t)N * K_S * sizeof(float), stream);
  (void)hipMemsetAsync(degb, 0, (size_t)N * sizeof(float), stream);

  {
    long long tot = (long long)N * 32;
    k_gather<<<(unsigned)((tot + 255) / 256), 256, 0, stream>>>(node_ids,
                                                                entity, hbuf, N);
  }
  {
    int tot = D_DIM * K_TOT;
    k_cvt_w<<<(tot + 255) / 256, 256, 0, stream>>>(weight, wself, WT);
  }
  {
    int tot = T_PAD * D_DIM;
    k_cvt_fc<<<(tot + 255) / 256, 256, 0, stream>>>(fc_w, FT);
  }
  {
    long long tot = (long long)E * 32;
    k_scatter<<<(unsigned)((tot + 255) / 256), 256, 0, stream>>>(
        src_idx, dst_idx, etype, hbuf, Sbuf, degb, E);
  }
  {
    int blocks = (N + MT - 1) / MT;  // N = 100000 -> exact multiple of 32
    k_rgcn_gemm<<<blocks, 256, 0, stream>>>(Sbuf, hbuf, degb, WT, FT, fc_b,
                                            outp, N);
  }
}
